// Model_59468117180501
// MI455X (gfx1250) — matrix operand order repaired and checked by reading, not compiled, not run on hardware
//
#include <hip/hip_runtime.h>
#include <hip/hip_bf16.h>
#include <math.h>

// ---------------------------------------------------------------------------
// Recurrent gated LM loss on MI455X (gfx1250, wave32, WMMA bf16 16x16x32).
//   P0  convert weights to bf16 (see split+transposed to [N][K]), zero state.
//   P1  gather embeddings (bf16, PAD row zeroed) -> X [S*B, F]
//   P2  GV = X @ see_v          (parallel WMMA GEMM, 16384x512x512)
//   P3  256x serial recurrence: G = state @ see_s; gate=sigmoid(G+GV+bias);
//       cont = state*gate + vecs*(1-gate)   (WMMA + elementwise, ping-pong)
//   P4  fused LN -> GEMM1(W_in,relu) -> GEMM2(W_out)  (h lives in 160KB LDS)
//   P5  logits GEMM vs emb^T, ONLINE softmax; emb vocab tiles double-buffered
//       into LDS with GLOBAL_LOAD_ASYNC_TO_LDS_B128 (ASYNCcnt pipeline),
//       shared by all 4 waves of the block (4x less L2 traffic).
//   P6  per-step masked mean, mean over steps -> scalar loss
// ---------------------------------------------------------------------------

#define Bb 64
#define Ss 256
#define Ff 512
#define Hh 2048
#define Vv 32000

typedef __attribute__((ext_vector_type(16))) __bf16 v16bf;
typedef __attribute__((ext_vector_type(8)))  __bf16 v8bf;
typedef __attribute__((ext_vector_type(8)))  float  v8f;

__device__ __forceinline__ v8f wmma_bf16(v16bf a, v16bf b, v8f c) {
  return __builtin_amdgcn_wmma_f32_16x16x32_bf16(false, a, false, b, (short)0, c,
                                                 false, false);
}

// A-matrix 16x32 (MxK) fragment: lane = M%16; lanes 0-15 hold K 0-7 & 16-23,
// lanes 16-31 hold K 8-15 & 24-31 (ISA 7.12.2). Row-major [M][K] source.
__device__ __forceinline__ v16bf load_a_frag(const __bf16* base, int ld, int m0,
                                             int k0, int lane) {
  int m = m0 + (lane & 15);
  int hi = lane >> 4;
  const __bf16* p = base + (size_t)m * ld + k0 + 8 * hi;
  v8bf lo = *(const v8bf*)p;
  v8bf hv = *(const v8bf*)(p + 16);
  return __builtin_shufflevector(lo, hv, 0, 1, 2, 3, 4, 5, 6, 7, 8, 9, 10, 11,
                                 12, 13, 14, 15);
}

// B-matrix 32x16 (KxN) fragment: lane = N%16; lanes 0-15 hold K 0-7 & 16-23,
// lanes 16-31 hold K 8-15 & 24-31. Source stored transposed: row-major [N][K].
__device__ __forceinline__ v16bf load_b_frag(const __bf16* baseT, int ld, int n0,
                                             int k0, int lane) {
  int n = n0 + (lane & 15);
  int hi = lane >> 4;
  const __bf16* p = baseT + (size_t)n * ld + k0 + 8 * hi;
  v8bf lo = *(const v8bf*)p;
  v8bf hv = *(const v8bf*)(p + 16);
  return __builtin_shufflevector(lo, hv, 0, 1, 2, 3, 4, 5, 6, 7, 8, 9, 10, 11,
                                 12, 13, 14, 15);
}

// Async DMA: global -> LDS, 16B per lane (ISA 10: GLOBAL_LOAD_ASYNC_TO_LDS_B128,
// tracked by ASYNCcnt). lds_addr = wave-relative LDS byte address.
__device__ __forceinline__ void async_g2l_b128(unsigned lds_addr,
                                               const void* gaddr) {
  asm volatile("global_load_async_to_lds_b128 %0, %1, off"
               :: "v"(lds_addr), "v"((unsigned long long)gaddr)
               : "memory");
}
__device__ __forceinline__ void wait_async0() {
  asm volatile("s_wait_asynccnt 0" ::: "memory");
}

// ---------------- P0: weight prep ----------------
__global__ void k_prep_see(const float* see, __bf16* seevT, __bf16* seesT) {
  int i = blockIdx.x * blockDim.x + threadIdx.x;  // i = f*512 + k
  int f = i >> 9, k = i & 511;
  seevT[i] = (__bf16)see[k * Ff + f];
  seesT[i] = (__bf16)see[(Ff + k) * Ff + f];
}

__global__ void k_cvt(const float* in, __bf16* out, int n) {
  int i = blockIdx.x * blockDim.x + threadIdx.x;
  if (i < n) out[i] = (__bf16)in[i];
}

__global__ void k_cvt_emb(const float* in, __bf16* out, int n) {
  int i = blockIdx.x * blockDim.x + threadIdx.x;
  if (i < n) out[i] = (i < Ff) ? (__bf16)0.0f : (__bf16)in[i];  // PAD row 0
}

__global__ void k_zero32(unsigned* p, int nwords) {
  int i = blockIdx.x * blockDim.x + threadIdx.x;
  if (i < nwords) p[i] = 0u;
}

// ---------------- P1: embedding gather ----------------
__global__ void k_gather(const int* inputs, const __bf16* embb, __bf16* X) {
  int r = blockIdx.x;                 // r = t*B + b
  int t = r >> 6, b = r & 63;
  int idx = inputs[b * Ss + t];
  const unsigned* src = (const unsigned*)(embb + (size_t)idx * Ff);
  unsigned* dst = (unsigned*)(X + (size_t)r * Ff);
  dst[threadIdx.x] = src[threadIdx.x];  // 256 threads x 4B = 1KB row
}

// ---------------- P2: GV = X @ see_v  (16384 x 512, K=512) ----------------
__global__ void k_gv(const __bf16* X, const __bf16* seevT, float* GV) {
  int gw = (blockIdx.x * blockDim.x + threadIdx.x) >> 5;  // 0..8191
  int lane = threadIdx.x & 31;
  int mt = gw >> 3;   // 1024 m-tiles
  int ng = gw & 7;    // 8 n-groups of 64
  int m0 = mt * 16;
  v8f acc[4] = {v8f{}, v8f{}, v8f{}, v8f{}};
  for (int k0 = 0; k0 < Ff; k0 += 32) {
    v16bf a = load_a_frag(X, Ff, m0, k0, lane);
#pragma unroll
    for (int j = 0; j < 4; j++) {
      v16bf b = load_b_frag(seevT, Ff, ng * 64 + j * 16, k0, lane);
      acc[j] = wmma_bf16(a, b, acc[j]);
    }
  }
  int hi = lane >> 4, nl = lane & 15;
#pragma unroll
  for (int j = 0; j < 4; j++)
#pragma unroll
    for (int v = 0; v < 8; v++)
      GV[(size_t)(m0 + v + 8 * hi) * Ff + ng * 64 + j * 16 + nl] = acc[j][v];
}

// ---------------- P3: serial recurrence step ----------------
__global__ void k_recur(const __bf16* stateb, const __bf16* seesT,
                        const float* gv_t, const float* bias, const float* prev,
                        const float* emb, const int* inputs, int t,
                        float* cont_out, __bf16* next_stateb) {
  int wave = (blockIdx.x * blockDim.x + threadIdx.x) >> 5;  // 0..127
  int lane = threadIdx.x & 31;
  int mt = wave >> 5, nt = wave & 31;  // 4 x 32 tiles of 16x16
  int m0 = mt * 16, n0 = nt * 16;
  v8f acc = {};
  for (int k0 = 0; k0 < Ff; k0 += 32) {
    v16bf a = load_a_frag(stateb, Ff, m0, k0, lane);
    v16bf b = load_b_frag(seesT, Ff, n0, k0, lane);
    acc = wmma_bf16(a, b, acc);
  }
  int hi = lane >> 4, n = n0 + (lane & 15);
  float bn = bias[n];
#pragma unroll
  for (int v = 0; v < 8; v++) {
    int m = m0 + v + 8 * hi;  // batch index
    float x = acc[v] + gv_t[m * Ff + n] + bn;
    float g = 1.0f / (1.0f + __expf(-x));
    int idx = inputs[m * Ss + t];
    float vec = (idx != 0) ? emb[(size_t)idx * Ff + n] : 0.0f;
    float c = prev[m * Ff + n] * g + vec * (1.0f - g);
    cont_out[m * Ff + n] = c;
    next_stateb[m * Ff + n] = (__bf16)c;
  }
}

// ------- P4: fused LayerNorm -> W_in/relu -> W_out (32 rows/block) -------
__global__ void k_mlp(const float* CONT, const __bf16* Winb, const __bf16* Woutb,
                      const float* ln_g, const float* ln_b, const float* b_in,
                      const float* b_out, __bf16* OUTb) {
  extern __shared__ char smem[];
  __bf16* xln = (__bf16*)smem;                      // 32 x 512  (32KB)
  __bf16* hbuf = (__bf16*)(smem + 32 * Ff * 2);     // 32 x 2048 (128KB)
  int wave = threadIdx.x >> 5, lane = threadIdx.x & 31;
  int r0 = blockIdx.x * 32;

  // LayerNorm: 8 rows per wave, lane-strided + shuffle reduction
  for (int rr = 0; rr < 8; rr++) {
    int row = wave * 8 + rr;
    const float* x = CONT + (size_t)(r0 + row) * Ff;
    float vals[16], s = 0.f, s2 = 0.f;
#pragma unroll
    for (int i = 0; i < 16; i++) {
      float v = x[lane + 32 * i];
      vals[i] = v; s += v; s2 += v * v;
    }
    for (int off = 16; off; off >>= 1) {
      s += __shfl_xor(s, off);
      s2 += __shfl_xor(s2, off);
    }
    float mu = s * (1.0f / Ff);
    float var = s2 * (1.0f / Ff) - mu * mu;
    float rstd = rsqrtf(var + 1e-5f);
#pragma unroll
    for (int i = 0; i < 16; i++) {
      int n = lane + 32 * i;
      xln[row * Ff + n] = (__bf16)((vals[i] - mu) * rstd * ln_g[n] + ln_b[n]);
    }
  }
  __syncthreads();

  // GEMM1: h = relu(xln @ W_in^T + b_in)   (2 x 128 tiles, K=512)
  for (int tl = wave; tl < 256; tl += 4) {
    int m0 = (tl & 1) * 16, n0 = (tl >> 1) * 16;
    v8f acc = {};
    for (int k0 = 0; k0 < Ff; k0 += 32) {
      v16bf a = load_a_frag(xln, Ff, m0, k0, lane);
      v16bf b = load_b_frag(Winb, Ff, n0, k0, lane);
      acc = wmma_bf16(a, b, acc);
    }
    int hi = lane >> 4, n = n0 + (lane & 15);
    float bi = b_in[n];
#pragma unroll
    for (int v = 0; v < 8; v++) {
      float h = acc[v] + bi;
      hbuf[(m0 + v + 8 * hi) * Hh + n] = (__bf16)(h > 0.f ? h : 0.f);
    }
  }
  __syncthreads();

  // GEMM2: out = h @ W_out^T + b_out       (2 x 32 tiles, K=2048)
  for (int tl = wave; tl < 64; tl += 4) {
    int m0 = (tl & 1) * 16, n0 = (tl >> 1) * 16;
    v8f acc = {};
    for (int k0 = 0; k0 < Hh; k0 += 32) {
      v16bf a = load_a_frag(hbuf, Hh, m0, k0, lane);
      v16bf b = load_b_frag(Woutb, Hh, n0, k0, lane);
      acc = wmma_bf16(a, b, acc);
    }
    int hi = lane >> 4, n = n0 + (lane & 15);
    float bo = b_out[n];
#pragma unroll
    for (int v = 0; v < 8; v++)
      OUTb[(size_t)(r0 + m0 + v + 8 * hi) * Ff + n] = (__bf16)(acc[v] + bo);
  }
}

// ------- P5: logits GEMM vs emb^T with online softmax, 64 rows/block -------
// 4 waves each own one 16x16 m-tile; vocab tiles (16 emb rows = 16KB) are
// double-buffered into LDS via async DMA and shared by all 4 waves.
// Dynamic LDS layout: [buf0 16KB][buf1 16KB][tgt 64*i32][tgtlog 64*f32]
#define LOG_LDS (16384 * 2 + 256 + 256)
__global__ void k_logits(const __bf16* OUTb, const __bf16* embb,
                         const int* targets, float* sumnll) {
  extern __shared__ char smem[];
  int* tgt = (int*)(smem + 32768);
  float* tgtlog = (float*)(smem + 32768 + 256);
  unsigned lds0 = (unsigned)(uintptr_t)(void*)smem;  // wave-relative LDS addr
  int wave = threadIdx.x >> 5, lane = threadIdx.x & 31;
  int tid = threadIdx.x;
  int r0 = blockIdx.x * 64;

  if (tid < 64) {
    int rg = r0 + tid;
    tgt[tid] = targets[(rg & 63) * Ss + (rg >> 6)];
    tgtlog[tid] = 0.f;
  }

  // Hoist this wave's A-fragments (16 rows x K=512 -> 16 frags, 128 VGPRs)
  int m0 = r0 + wave * 16;
  v16bf afr[16];
#pragma unroll
  for (int kk = 0; kk < 16; kk++)
    afr[kk] = load_a_frag(OUTb, Ff, m0, kk * 32, lane);

  float rmax[8], rsum[8];
#pragma unroll
  for (int v = 0; v < 8; v++) { rmax[v] = -1e30f; rsum[v] = 0.f; }
  int hi = lane >> 4, nl = lane & 15;
  const float inv = 0.044194173824159216f;  // 1/sqrt(512)

  // prologue: stage vocab tile 0 into buf0 (8 x b128 per thread = 16KB/block)
  {
    const char* g = (const char*)embb + tid * 16;
    unsigned l = lds0 + tid * 16;
#pragma unroll
    for (int i = 0; i < 8; i++) async_g2l_b128(l + i * 2048, g + i * 2048);
  }

  for (int vt = 0; vt < Vv / 16; vt++) {   // 2000 vocab tiles
    int cur = vt & 1;
    wait_async0();        // this wave's staged data has landed in LDS
    __syncthreads();      // all waves' stages landed; prior reads finished
    if (vt + 1 < Vv / 16) {  // stage next tile while computing this one
      const char* g = (const char*)(embb + (size_t)(vt + 1) * 16 * Ff) + tid * 16;
      unsigned l = lds0 + (unsigned)((cur ^ 1) * 16384) + tid * 16;
#pragma unroll
      for (int i = 0; i < 8; i++) async_g2l_b128(l + i * 2048, g + i * 2048);
    }
    const __bf16* bb = (const __bf16*)(smem + cur * 16384);
    v8f acc = {};
#pragma unroll
    for (int kk = 0; kk < 16; kk++) {
      v16bf b = load_b_frag(bb, Ff, 0, kk * 32, lane);
      acc = wmma_bf16(afr[kk], b, acc);
    }
    int col = vt * 16 + nl;
#pragma unroll
    for (int v = 0; v < 8; v++) {
      float x = acc[v] * inv;
      int row = wave * 16 + v + 8 * hi;
      if (col == tgt[row]) tgtlog[row] = x;  // unique writer per row
      float nm = fmaxf(rmax[v], x);
      rsum[v] = rsum[v] * __expf(rmax[v] - nm) + __expf(x - nm);
      rmax[v] = nm;
    }
  }
  __syncthreads();

  // Per-row merge across the 16 lanes of each half-wave (butterfly shuffles)
#pragma unroll
  for (int v = 0; v < 8; v++) {
    float gm = rmax[v];
#pragma unroll
    for (int mk = 8; mk; mk >>= 1) gm = fmaxf(gm, __shfl_xor(gm, mk));
    float y = rsum[v] * __expf(rmax[v] - gm);
#pragma unroll
    for (int mk = 8; mk; mk >>= 1) y += __shfl_xor(y, mk);
    if (nl == 0) {
      int row = wave * 16 + v + 8 * hi;
      float nll = (gm + __logf(y)) - tgtlog[row];
      if (tgt[row] != 0) atomicAdd(&sumnll[(r0 + row) >> 6], nll);
    }
  }
}

// ---------------- P6: finalize ----------------
__global__ void k_final(const float* sumnll, const int* targets, float* out) {
  __shared__ float red[256];
  int t = threadIdx.x;
  int cnt = 0;
  for (int b = 0; b < Bb; b++) cnt += (targets[b * Ss + t] != 0);
  red[t] = sumnll[t] / fmaxf((float)cnt, 1.0f);
  __syncthreads();
  for (int off = 128; off; off >>= 1) {
    if (t < off) red[t] += red[t + off];
    __syncthreads();
  }
  if (t == 0) out[0] = red[0] * (1.0f / Ss);
}

// ---------------------------------------------------------------------------
extern "C" void kernel_launch(void* const* d_in, const int* in_sizes, int n_in,
                              void* d_out, int out_size, void* d_ws,
                              size_t ws_size, hipStream_t stream) {
  const int* inputs = (const int*)d_in[0];
  const int* targets = (const int*)d_in[1];
  const float* see = (const float*)d_in[2];
  const float* bias = (const float*)d_in[3];
  const float* W_in = (const float*)d_in[4];
  const float* b_in = (const float*)d_in[5];
  const float* W_out = (const float*)d_in[6];
  const float* b_out = (const float*)d_in[7];
  const float* ln_g = (const float*)d_in[8];
  const float* ln_b = (const float*)d_in[9];
  const float* emb = (const float*)d_in[10];
  float* out = (float*)d_out;

  // workspace carve (all sizes 256B-multiples; zero-region kept contiguous)
  char* p = (char*)d_ws;
  __bf16* embb  = (__bf16*)p; p += (size_t)Vv * Ff * 2;          // 32.77MB
  __bf16* seevT = (__bf16*)p; p += (size_t)Ff * Ff * 2;
  __bf16* seesT = (__bf16*)p; p += (size_t)Ff * Ff * 2;
  __bf16* Winb  = (__bf16*)p; p += (size_t)Hh * Ff * 2;
  __bf16* Woutb = (__bf16*)p; p += (size_t)Ff * Hh * 2;
  __bf16* X     = (__bf16*)p; p += (size_t)Ss * Bb * Ff * 2;     // 16.78MB
  float*  GV    = (float*)p;  p += (size_t)Ss * Bb * Ff * 4;     // 33.55MB
  float*  CONT  = (float*)p;  p += (size_t)Ss * Bb * Ff * 4;     // 33.55MB
  __bf16* OUTb  = (__bf16*)p; p += (size_t)Ss * Bb * Ff * 2;     // 16.78MB
  float*  st0   = (float*)p;  p += (size_t)Bb * Ff * 4;          // zeroed
  __bf16* stA   = (__bf16*)p; p += (size_t)Bb * Ff * 2;          // zeroed
  __bf16* stB   = (__bf16*)p; p += (size_t)Bb * Ff * 2;          // zeroed
  float*  SUMNLL = (float*)p; p += (size_t)Ss * 4;               // zeroed

  // P0: weight prep + zero (zero-region: st0,stA,stB,SUMNLL contiguous)
  k_prep_see<<<1024, 256, 0, stream>>>(see, seevT, seesT);
  k_cvt<<<4096, 256, 0, stream>>>(W_in, Winb, Hh * Ff);
  k_cvt<<<4096, 256, 0, stream>>>(W_out, Woutb, Ff * Hh);
  k_cvt_emb<<<Vv * Ff / 256, 256, 0, stream>>>(emb, embb, Vv * Ff);
  int zw = (Bb * Ff * 4 + Bb * Ff * 2 * 2 + Ss * 4) / 4;  // 65792 words
  k_zero32<<<(zw + 255) / 256, 256, 0, stream>>>((unsigned*)st0, zw);

  // P1 + P2
  k_gather<<<Ss * Bb, 256, 0, stream>>>(inputs, embb, X);
  k_gv<<<2048, 128, 0, stream>>>(X, seevT, GV);

  // P3: serial recurrence, ping-pong bf16 state
  for (int t = 0; t < Ss; t++) {
    const __bf16* sb = (t & 1) ? stB : stA;
    __bf16* nb = (t & 1) ? stA : stB;
    const float* prev = (t == 0) ? st0 : (CONT + (size_t)(t - 1) * Bb * Ff);
    k_recur<<<32, 128, 0, stream>>>(sb, seesT, GV + (size_t)t * Bb * Ff, bias,
                                    prev, emb, inputs, t,
                                    CONT + (size_t)t * Bb * Ff, nb);
  }

  // P4: fused MLP over all 16384 rows (160KB dynamic LDS per block)
  size_t lds = (size_t)32 * Ff * 2 + (size_t)32 * Hh * 2;  // 163840B
  k_mlp<<<Ss * Bb / 32, 128, lds, stream>>>(CONT, Winb, Woutb, ln_g, ln_b, b_in,
                                            b_out, OUTb);

  // P5 + P6
  k_logits<<<Ss * Bb / 16 / 4, 128, LOG_LDS, stream>>>(OUTb, embb, targets,
                                                       SUMNLL);
  k_final<<<1, 256, 0, stream>>>(SUMNLL, targets, out);
}
